// MultiRelationGATBlock_58626303591101
// MI455X (gfx1250) — compile-verified, matrix operand order checked by reading, not run on hardware
//
#include <hip/hip_runtime.h>
#include <hip/hip_bf16.h>

// ---------------- problem constants (from reference) ----------------
#define NN 50000
#define EE 400000
#define HH 64
#define RR 4
#define HEADS 4
#define OUTC 64
#define HO 256            // HEADS*OUT
#define EA 16
#define REL 8
#define ED 24             // EA+REL
#define FFN 128
#define KB 8              // spline bases
#define K1 576            // 64 + 64*8  (fused KAN1 inner dim)
#define K2 1152           // 128 + 128*8 (fused KAN2 inner dim)

typedef _Float16 v16h __attribute__((ext_vector_type(16)));
typedef _Float16 v8h  __attribute__((ext_vector_type(8)));
typedef float    v8f  __attribute__((ext_vector_type(8)));

// ---------------- WMMA fragment helpers (CDNA5 16-bit layout) ----------------
// A 16x32 (or B^T treated identically): lane L holds row (L&15);
// halves 0..7  = K (L<16 ? 0 : 8)+0..7, halves 8..15 = same +16.
__device__ __forceinline__ v16h load_frag(const _Float16* base, int ld, int lane) {
  int row = lane & 15;
  int kb  = (lane >> 4) << 3;
  const _Float16* p = base + (size_t)row * ld + kb;
  union { v16h v; v8h h[2]; } u;
  u.h[0] = *(const v8h*)(p);
  u.h[1] = *(const v8h*)(p + 16);
  return u.v;
}

__device__ __forceinline__ v8f wmma_f16(v16h a, v16h b, v8f c) {
  return __builtin_amdgcn_wmma_f32_16x16x32_f16(false, a, false, b, (short)0, c, false, false);
}

// ---------------- float atomic-max via order-preserving uint key ----------------
__device__ __forceinline__ void atomicMaxFloatKey(unsigned* addr, float v) {
  unsigned uv = __float_as_uint(v);
  unsigned key = (uv & 0x80000000u) ? ~uv : (uv | 0x80000000u);
  atomicMax(addr, key);
}
__device__ __forceinline__ float decodeKey(unsigned key) {
  unsigned uv = (key & 0x80000000u) ? (key & 0x7fffffffu) : ~key;
  return __uint_as_float(uv);
}

// ---------------- cubic B-spline bases on the uniform grid ----------------
// grid point g(t) = 0.4*t - 2.2, t = 0..11 ; returns 8 order-3 bases.
__device__ __forceinline__ void bspline8(float x, float* o) {
  float b0[11];
#pragma unroll
  for (int t = 0; t < 11; ++t) {
    float gl = 0.4f * t - 2.2f;
    b0[t] = (x >= gl && x < gl + 0.4f) ? 1.0f : 0.0f;
  }
  float b1[10];
#pragma unroll
  for (int i = 0; i < 10; ++i) {
    float gi = 0.4f * i - 2.2f;
    b1[i] = (x - gi) * 2.5f * b0[i] + ((gi + 0.8f) - x) * 2.5f * b0[i + 1];
  }
  float b2[9];
#pragma unroll
  for (int i = 0; i < 9; ++i) {
    float gi = 0.4f * i - 2.2f;
    b2[i] = (x - gi) * 1.25f * b1[i] + ((gi + 1.2f) - x) * 1.25f * b1[i + 1];
  }
#pragma unroll
  for (int i = 0; i < 8; ++i) {
    float gi = 0.4f * i - 2.2f;
    o[i] = (x - gi) * (1.0f / 1.2f) * b2[i] + ((gi + 1.6f) - x) * (1.0f / 1.2f) * b2[i + 1];
  }
}

__device__ __forceinline__ float silu(float x) { return x / (1.0f + __expf(-x)); }

// ==================== prep kernels ====================
__global__ void k_h2f16(const float* h, _Float16* h16) {
  int i = blockIdx.x * blockDim.x + threadIdx.x;
  if (i < NN * HH) h16[i] = (_Float16)h[i];
}

// wT layout: [side(2)][r(4)][col(256)][k(64)] f16, col-major weights for B frags
__global__ void k_wT(const float* Wl, const float* Wr, _Float16* wT) {
  int idx = blockIdx.x * blockDim.x + threadIdx.x;
  if (idx >= 2 * RR * HO * HH) return;
  int k = idx & 63;
  int col = (idx >> 6) & 255;
  int r = (idx >> 14) & 3;
  int side = idx >> 16;
  const float* W = side ? Wr : Wl;           // [R][64][256]
  wT[idx] = (_Float16)W[((size_t)r * HH + k) * HO + col];
}

// fused KAN weights, col-major: W1t[o][k] (128x576), W2t[o][k] (64x1152)
__global__ void k_kanw(const float* bw1, const float* sw1, const float* sc1,
                       const float* bw2, const float* sw2, const float* sc2,
                       _Float16* w1t, _Float16* w2t) {
  int idx = blockIdx.x * blockDim.x + threadIdx.x;
  if (idx < FFN * K1) {
    int o = idx / K1, k = idx % K1;
    float v;
    if (k < HH) v = bw1[o * HH + k];
    else {
      int i = (k - HH) >> 3, kb = (k - HH) & 7;
      v = sw1[((size_t)o * HH + i) * KB + kb] * sc1[o * HH + i];
    }
    w1t[idx] = (_Float16)v;
  } else if (idx < FFN * K1 + HH * K2) {
    int j = idx - FFN * K1;
    int o = j / K2, k = j % K2;
    float v;
    if (k < FFN) v = bw2[o * FFN + k];
    else {
      int i = (k - FFN) >> 3, kb = (k - FFN) & 7;
      v = sw2[((size_t)o * FFN + i) * KB + kb] * sc2[o * FFN + i];
    }
    w2t[j] = (_Float16)v;
  }
}

__global__ void k_gate_bias(const float* rel_gate, const float* out_bias,
                            float* gate, float* biasvec) {
  int t = threadIdx.x;  // 64 threads
  float g0 = rel_gate[0], g1 = rel_gate[1], g2 = rel_gate[2], g3 = rel_gate[3];
  float mx = fmaxf(fmaxf(g0, g1), fmaxf(g2, g3));
  float e0 = __expf(g0 - mx), e1 = __expf(g1 - mx);
  float e2 = __expf(g2 - mx), e3 = __expf(g3 - mx);
  float s = e0 + e1 + e2 + e3;
  float gv0 = e0 / s, gv1 = e1 / s, gv2 = e2 / s, gv3 = e3 / s;
  if (t == 0) { gate[0] = gv0; gate[1] = gv1; gate[2] = gv2; gate[3] = gv3; }
  float b = gv0 * out_bias[0 * OUTC + t] + gv1 * out_bias[1 * OUTC + t] +
            gv2 * out_bias[2 * OUTC + t] + gv3 * out_bias[3 * OUTC + t];
  biasvec[t] = b;
}

// ==================== xl/xr node GEMM (WMMA) ====================
// grid.x = 3125 row tiles, grid.y = 16 : bit0=side, bits1..2=r, bit3=col half
__global__ void k_node_gemm(const _Float16* h16, const _Float16* wT,
                            const float* bl, const float* br,
                            _Float16* xl, _Float16* xr) {
  int wave = threadIdx.x >> 5, lane = threadIdx.x & 31;
  int row0 = blockIdx.x * 16;
  int yy = blockIdx.y;
  int side = yy & 1, r = (yy >> 1) & 3, half = yy >> 3;
  int col0 = half * 128 + wave * 16;
  const _Float16* A = h16 + (size_t)row0 * HH;
  const _Float16* B = wT + (((size_t)side * RR + r) * HO + col0) * HH;
  v8f acc = {};
#pragma unroll
  for (int k0 = 0; k0 < HH; k0 += 32) {
    v16h a = load_frag(A + k0, HH, lane);
    v16h b = load_frag(B + k0, HH, lane);
    acc = wmma_f16(a, b, acc);
  }
  const float* bias = side ? br : bl;
  _Float16* outp = side ? xr : xl;
  int col = col0 + (lane & 15);
  float bv = bias[r * HO + col];
  int rbase = row0 + ((lane >> 4) << 3);
#pragma unroll
  for (int i = 0; i < 8; ++i)
    outp[((size_t)r * NN + rbase + i) * HO + col] = (_Float16)(acc[i] + bv);
}

// ==================== edge pass 1: logits + segment max ====================
__global__ void k_edge_logits(const int* ei, const int* et, const float* edge_attr,
                              const float* rel_emb, const float* We, const float* be,
                              const float* att, const _Float16* xl, const _Float16* xr,
                              float* logits, unsigned* mkeys) {
  int wid = (blockIdx.x * blockDim.x + threadIdx.x) >> 5;
  int lane = threadIdx.x & 31;
  if (wid >= EE) return;
  int e = wid;
  int r = et[e];
  int src = ei[e], dst = ei[EE + e];
  float ea[ED];
#pragma unroll
  for (int i = 0; i < EA; ++i) ea[i] = edge_attr[(size_t)e * EA + i];
#pragma unroll
  for (int i = 0; i < REL; ++i) ea[EA + i] = rel_emb[r * REL + i];
  int head = lane >> 3;
  const float* Wr_ = We + (size_t)r * ED * HO;
  const _Float16* xls = xl + ((size_t)r * NN + src) * HO;
  const _Float16* xrd = xr + ((size_t)r * NN + dst) * HO;
  float part = 0.f;
#pragma unroll
  for (int j = 0; j < 8; ++j) {
    int col = lane * 8 + j;
    float ev = be[r * HO + col];
#pragma unroll
    for (int i = 0; i < ED; ++i) ev = fmaf(ea[i], Wr_[i * HO + col], ev);
    float z = (float)xls[col] + (float)xrd[col] + ev;
    z = z > 0.f ? z : 0.2f * z;               // leaky_relu(0.2)
    part = fmaf(z, att[(r * HEADS + head) * OUTC + (col & 63)], part);
  }
  part += __shfl_xor(part, 1, 32);
  part += __shfl_xor(part, 2, 32);
  part += __shfl_xor(part, 4, 32);
  if ((lane & 7) == 0) {
    logits[(size_t)e * HEADS + head] = part;
    atomicMaxFloatKey(&mkeys[((size_t)r * NN + dst) * HEADS + head], part);
  }
}

// ==================== edge pass 2: exp + denom ====================
__global__ void k_edge_softmax(const int* ei, const int* et, const unsigned* mkeys,
                               float* logits, float* denom) {
  int idx = blockIdx.x * blockDim.x + threadIdx.x;
  if (idx >= EE * HEADS) return;
  int e = idx >> 2, hh = idx & 3;
  int r = et[e], dst = ei[EE + e];
  float mm = decodeKey(mkeys[((size_t)r * NN + dst) * HEADS + hh]);
  float ex = __expf(logits[idx] - mm);
  logits[idx] = ex;
  atomicAdd(&denom[((size_t)r * NN + dst) * HEADS + hh], ex);
}

// ==================== edge pass 3: alpha-weighted head-mean scatter ====================
__global__ void k_edge_aggr(const int* ei, const int* et, const float* logits,
                            const float* denom, const _Float16* xl,
                            const float* gate, float* hmsg) {
  int wid = (blockIdx.x * blockDim.x + threadIdx.x) >> 5;
  int lane = threadIdx.x & 31;
  if (wid >= EE) return;
  int e = wid;
  int r = et[e], src = ei[e], dst = ei[EE + e];
  float gr = gate[r] * 0.25f;                 // fold mean over 4 heads
  float al[HEADS];
#pragma unroll
  for (int h = 0; h < HEADS; ++h)
    al[h] = logits[(size_t)e * HEADS + h] /
            (denom[((size_t)r * NN + dst) * HEADS + h] + 1e-16f);
  const _Float16* xs = xl + ((size_t)r * NN + src) * HO;
  float v0 = 0.f, v1 = 0.f;
#pragma unroll
  for (int h = 0; h < HEADS; ++h) {
    v0 = fmaf(al[h], (float)xs[h * OUTC + lane], v0);
    v1 = fmaf(al[h], (float)xs[h * OUTC + lane + 32], v1);
  }
  atomicAdd(&hmsg[(size_t)dst * HH + lane], gr * v0);
  atomicAdd(&hmsg[(size_t)dst * HH + lane + 32], gr * v1);
}

// ==================== LayerNorm (wave per node, 64 ch) ====================
__global__ void k_ln(const float* a, const float* bsrc, const float* bias64,
                     const float* g, const float* bp, float* out) {
  int wid = (blockIdx.x * blockDim.x + threadIdx.x) >> 5;
  int lane = threadIdx.x & 31;
  if (wid >= NN) return;
  size_t base = (size_t)wid * HH;
  float add0 = bias64 ? bias64[lane] : 0.f;
  float add1 = bias64 ? bias64[lane + 32] : 0.f;
  float v0 = a[base + lane] + bsrc[base + lane] + add0;
  float v1 = a[base + lane + 32] + bsrc[base + lane + 32] + add1;
  float s = v0 + v1;
#pragma unroll
  for (int m = 1; m < 32; m <<= 1) s += __shfl_xor(s, m, 32);
  float mu = s * (1.0f / 64.0f);
  float d0 = v0 - mu, d1 = v1 - mu;
  float q = d0 * d0 + d1 * d1;
#pragma unroll
  for (int m = 1; m < 32; m <<= 1) q += __shfl_xor(q, m, 32);
  float rstd = rsqrtf(q * (1.0f / 64.0f) + 1e-5f);
  out[base + lane]      = d0 * rstd * g[lane] + bp[lane];
  out[base + lane + 32] = d1 * rstd * g[lane + 32] + bp[lane + 32];
}

// ==================== KAN layer 1: LDS-fused spline expansion + WMMA ====================
__global__ void k_kan1(const float* x, const _Float16* w1t, float* y1) {
  __shared__ __align__(16) _Float16 As[16 * K1];   // 18 KB
  int row0 = blockIdx.x * 16;
  for (int p = threadIdx.x; p < 16 * HH; p += 256) {
    int rr = p >> 6, i = p & 63;
    float xv = x[((size_t)row0 + rr) * HH + i];
    As[rr * K1 + i] = (_Float16)silu(xv);
    float bs[KB]; bspline8(xv, bs);
#pragma unroll
    for (int k = 0; k < KB; ++k) As[rr * K1 + HH + i * KB + k] = (_Float16)bs[k];
  }
  __syncthreads();
  int wave = threadIdx.x >> 5, lane = threadIdx.x & 31;
  int col0 = wave * 16;                            // 8 waves -> 128 cols
  v8f acc = {};
  for (int k0 = 0; k0 < K1; k0 += 32) {
    v16h a = load_frag(&As[k0], K1, lane);
    v16h b = load_frag(w1t + (size_t)col0 * K1 + k0, K1, lane);
    acc = wmma_f16(a, b, acc);
  }
  int col = col0 + (lane & 15);
  int rbase = row0 + ((lane >> 4) << 3);
#pragma unroll
  for (int i = 0; i < 8; ++i)
    y1[((size_t)(rbase + i)) * FFN + col] = acc[i];
}

// ==================== KAN layer 2: LDS-fused spline expansion + WMMA ====================
__global__ void k_kan2(const float* y1, const _Float16* w2t, float* y2) {
  __shared__ __align__(16) _Float16 As[16 * K2];   // 36 KB
  int row0 = blockIdx.x * 16;
  for (int p = threadIdx.x; p < 16 * FFN; p += 128) {
    int rr = p >> 7, j = p & 127;
    float xv = y1[((size_t)row0 + rr) * FFN + j];
    As[rr * K2 + j] = (_Float16)silu(xv);
    float bs[KB]; bspline8(xv, bs);
#pragma unroll
    for (int k = 0; k < KB; ++k) As[rr * K2 + FFN + j * KB + k] = (_Float16)bs[k];
  }
  __syncthreads();
  int wave = threadIdx.x >> 5, lane = threadIdx.x & 31;
  int col0 = wave * 16;                            // 4 waves -> 64 cols
  v8f acc = {};
  for (int k0 = 0; k0 < K2; k0 += 32) {
    v16h a = load_frag(&As[k0], K2, lane);
    v16h b = load_frag(w2t + (size_t)col0 * K2 + k0, K2, lane);
    acc = wmma_f16(a, b, acc);
  }
  int col = col0 + (lane & 15);
  int rbase = row0 + ((lane >> 4) << 3);
#pragma unroll
  for (int i = 0; i < 8; ++i)
    y2[((size_t)(rbase + i)) * HH + col] = acc[i];
}

// ==================== host launch ====================
extern "C" void kernel_launch(void* const* d_in, const int* in_sizes, int n_in,
                              void* d_out, int out_size, void* d_ws, size_t ws_size,
                              hipStream_t stream) {
  (void)in_sizes; (void)n_in; (void)out_size; (void)ws_size;
  const float* h         = (const float*)d_in[0];
  const int*   ei        = (const int*)d_in[1];
  const float* edge_attr = (const float*)d_in[2];
  const int*   et        = (const int*)d_in[3];
  const float* rel_emb   = (const float*)d_in[4];
  const float* rel_gate  = (const float*)d_in[5];
  const float* Wl        = (const float*)d_in[6];
  const float* bl        = (const float*)d_in[7];
  const float* Wr        = (const float*)d_in[8];
  const float* br        = (const float*)d_in[9];
  const float* We        = (const float*)d_in[10];
  const float* be        = (const float*)d_in[11];
  const float* att       = (const float*)d_in[12];
  const float* out_bias  = (const float*)d_in[13];
  const float* ln1_g     = (const float*)d_in[14];
  const float* ln1_b     = (const float*)d_in[15];
  const float* ln2_g     = (const float*)d_in[16];
  const float* ln2_b     = (const float*)d_in[17];
  const float* base_w1   = (const float*)d_in[18];
  const float* spline_w1 = (const float*)d_in[19];
  const float* scaler1   = (const float*)d_in[20];
  const float* base_w2   = (const float*)d_in[22];
  const float* spline_w2 = (const float*)d_in[23];
  const float* scaler2   = (const float*)d_in[24];

  char* w = (char*)d_ws;
  size_t off = 0;
  auto alloc = [&](size_t bytes) -> void* {
    off = (off + 255) & ~(size_t)255;
    void* p = w + off;
    off += bytes;
    return p;
  };
  _Float16* h16     = (_Float16*)alloc((size_t)NN * HH * 2);
  _Float16* xl      = (_Float16*)alloc((size_t)RR * NN * HO * 2);
  _Float16* xr      = (_Float16*)alloc((size_t)RR * NN * HO * 2);
  _Float16* wT      = (_Float16*)alloc((size_t)2 * RR * HO * HH * 2);
  _Float16* w1t     = (_Float16*)alloc((size_t)FFN * K1 * 2);
  _Float16* w2t     = (_Float16*)alloc((size_t)HH * K2 * 2);
  float*    gate    = (float*)alloc(RR * 4);
  float*    biasvec = (float*)alloc(HH * 4);
  unsigned* mkeys   = (unsigned*)alloc((size_t)RR * NN * HEADS * 4);
  float*    denom   = (float*)alloc((size_t)RR * NN * HEADS * 4);
  float*    logits  = (float*)alloc((size_t)EE * HEADS * 4);
  float*    hmsg    = (float*)alloc((size_t)NN * HH * 4);
  float*    xbuf    = (float*)alloc((size_t)NN * HH * 4);
  float*    y1      = (float*)alloc((size_t)NN * FFN * 4);
  float*    y2      = (float*)alloc((size_t)NN * HH * 4);

  // prep
  k_h2f16<<<(NN * HH + 255) / 256, 256, 0, stream>>>(h, h16);
  k_wT<<<(2 * RR * HO * HH + 255) / 256, 256, 0, stream>>>(Wl, Wr, wT);
  k_kanw<<<(FFN * K1 + HH * K2 + 255) / 256, 256, 0, stream>>>(
      base_w1, spline_w1, scaler1, base_w2, spline_w2, scaler2, w1t, w2t);
  k_gate_bias<<<1, 64, 0, stream>>>(rel_gate, out_bias, gate, biasvec);
  hipMemsetAsync(mkeys, 0, (size_t)RR * NN * HEADS * 4, stream);
  hipMemsetAsync(denom, 0, (size_t)RR * NN * HEADS * 4, stream);
  hipMemsetAsync(hmsg, 0, (size_t)NN * HH * 4, stream);

  // per-relation node transforms (WMMA)
  k_node_gemm<<<dim3(NN / 16, 16), 256, 0, stream>>>(h16, wT, bl, br, xl, xr);

  // edge passes
  k_edge_logits<<<EE / 8, 256, 0, stream>>>(ei, et, edge_attr, rel_emb, We, be,
                                            att, xl, xr, logits, mkeys);
  k_edge_softmax<<<(EE * HEADS + 255) / 256, 256, 0, stream>>>(ei, et, mkeys,
                                                               logits, denom);
  k_edge_aggr<<<EE / 8, 256, 0, stream>>>(ei, et, logits, denom, xl, gate, hmsg);

  // LN1: x = LN(h + hmsg + sum_r gate_r*out_bias_r)
  k_ln<<<(NN + 7) / 8, 256, 0, stream>>>(h, hmsg, biasvec, ln1_g, ln1_b, xbuf);

  // KAN FFN (WMMA with LDS-built spline expansion)
  k_kan1<<<NN / 16, 256, 0, stream>>>(xbuf, w1t, y1);
  k_kan2<<<NN / 16, 128, 0, stream>>>(y1, w2t, y2);

  // LN2 -> output
  k_ln<<<(NN + 7) / 8, 256, 0, stream>>>(xbuf, y2, (const float*)nullptr,
                                         ln2_g, ln2_b, (float*)d_out);
}
